// OctreeCrossAttention_56607668961509
// MI455X (gfx1250) — matrix operand order repaired and checked by reading, not compiled, not run on hardware
//
#include <hip/hip_runtime.h>
#include <hip/hip_bf16.h>
#include <cstdint>

// ---------------- problem constants (match reference) ----------------
#define NPTS    131072          // N
#define DIMC    256             // DIM
#define NHEADS  8
#define PK      32              // patch size K
#define HDIM    32              // head dim
#define MPATCH  (NPTS / PK)     // 4096
#define POSB    25
#define RPEN    51              // 2*POSB+1
#define RPE3    (3 * RPEN)      // 153
#define SCALE_F 0.17677669529663687f   // 32^-0.5

typedef __attribute__((ext_vector_type(16))) _Float16 v16h;
typedef __attribute__((ext_vector_type(8)))  float    v8f;
typedef __attribute__((ext_vector_type(4)))  unsigned u32x4;
typedef __attribute__((ext_vector_type(8)))  int      i32x8;
typedef __attribute__((ext_vector_type(4)))  int      i32x4;

union Frag16 { v16h h; unsigned u[8]; _Float16 e[16]; };
union H8     { uint4 u4; _Float16 h[8]; };
union H2     { unsigned u; _Float16 h[2]; };

struct __align__(4) I3 { int x, y, z; };   // 12B -> global_load_b96

#if __has_builtin(__builtin_amdgcn_tensor_load_to_lds)
#define HAS_TDM 1
#endif

__device__ inline v8f v8f_zero() {
    v8f z;
#pragma unroll
    for (int i = 0; i < 8; ++i) z[i] = 0.0f;
    return z;
}

__device__ inline v8f wmma32(v16h a, v16h b, v8f c) {
    return __builtin_amdgcn_wmma_f32_16x16x32_f16(
        false, a, false, b, (short)0, c, false, false);
}

// async global->LDS copy of 64 contiguous bytes (one 32-half tile row).
// INST_OFFSET is added to BOTH the LDS and global address per ISA 08 4.4.
__device__ inline void async_row64(unsigned lds_off, unsigned long long gaddr) {
    asm volatile(
        "global_load_async_to_lds_b128 %0, %1, off\n\t"
        "global_load_async_to_lds_b128 %0, %1, off offset:16\n\t"
        "global_load_async_to_lds_b128 %0, %1, off offset:32\n\t"
        "global_load_async_to_lds_b128 %0, %1, off offset:48"
        :: "v"(lds_off), "v"(gaddr) : "memory");
}
__device__ inline void async_wait0() {
    asm volatile("s_wait_asynccnt 0x0" ::: "memory");
}

// A fragment (16x32 f16), tile row-major [row][k], pitch halves (even).
__device__ inline v16h frag_a_rowmajor(const _Float16* tile, int pitch, int lane) {
    Frag16 f;
    const unsigned* rowp = (const unsigned*)(tile + (lane & 15) * pitch);
    const int kb = (lane & 16) ? 8 : 0;
#pragma unroll
    for (int v = 0; v < 8; ++v) f.u[v] = rowp[kb + v];
    return f.h;
}

// B fragment (32x16 f16) from K-contiguous storage tileT[n][k], pitch halves (even).
__device__ inline v16h frag_b_kcontig(const _Float16* tileT, int pitch, int lane) {
    Frag16 f;
    const unsigned* colp = (const unsigned*)(tileT + (lane & 15) * pitch);
    const int kb = (lane & 16) ? 8 : 0;
#pragma unroll
    for (int v = 0; v < 8; ++v) f.u[v] = colp[kb + v];
    return f.h;
}

// B fragment (32x16 f16) from row-major [k][n] storage (strided u16 gathers).
__device__ inline v16h frag_b_nmajor(const _Float16* tile, int pitch, int lane) {
    Frag16 f;
    const int n  = lane & 15;
    const int kb = (lane & 16) ? 16 : 0;
#pragma unroll
    for (int v = 0; v < 8; ++v) {
        f.e[2 * v]     = tile[(kb + 2 * v)     * pitch + n];
        f.e[2 * v + 1] = tile[(kb + 2 * v + 1) * pitch + n];
    }
    return f.h;
}

// ---------------- tiled WMMA GEMM: out[rows x P] = A[rows x 256] * W[256 x P] + bias ----------------
// 256 threads = 8 waves (4x2), block tile 128x64, wave tile 32x32, k-step 32.
// A_F16 path stages the A tile with one TDM tensor_load_to_lds (padded to pitch 40).
template <bool A_F16, bool OUT_F32>
__global__ __launch_bounds__(256) void gemm_kernel(const void* __restrict__ A_,
                                                   const float* __restrict__ W,
                                                   const float* __restrict__ bias,
                                                   void* __restrict__ out_, int P) {
    __shared__ __align__(16) _Float16 a_lds[128 * 40];   // 128x32, pitch 40
    __shared__ __align__(16) _Float16 bt_lds[64 * 40];   // transposed: [col][k], pitch 40

    const int tid  = threadIdx.x;
    const int lane = tid & 31;
    const int wave = tid >> 5;
    const long rbase = (long)blockIdx.y * 128;
    const int  cbase = blockIdx.x * 64;
    const int  wr = (wave >> 1) * 32;
    const int  wc = (wave & 1) * 32;

    v8f c[2][2];
#pragma unroll
    for (int i = 0; i < 2; ++i)
#pragma unroll
        for (int j = 0; j < 2; ++j) c[i][j] = v8f_zero();

    for (int kc = 0; kc < DIMC; kc += 32) {
        // ---- stage A tile 128x32 ----
        if constexpr (A_F16) {
#ifdef HAS_TDM
            if (wave == 0) {
                // Tensor DMA descriptor (ISA 08 §8.3-8.5): 2D tile, 2B elems,
                // dim0=32 (contig), dim1=128 rows, row stride 256 halves,
                // LDS pad: 16B after every 64B -> pitch 40 halves.
                const unsigned long long ga =
                    (unsigned long long)(uintptr_t)((const _Float16*)A_ + rbase * DIMC + kc);
                const unsigned lds = (unsigned)(uintptr_t)&a_lds[0];
                u32x4 g0 = {1u /*count=1*/, lds,
                            (unsigned)ga, (unsigned)(ga >> 32) | 0x80000000u /*type=2*/};
                i32x8 g1 = {0x06D10000,       // data_size=2B, pad_en, interval=64B, amount=16B
                            32 << 16,         // tensor_dim0 = 32
                            128 << 16,        // tensor_dim1 = 128
                            32 << 16,         // tile_dim0 = 32
                            128,              // tile_dim1 = 128
                            256,              // tensor_dim0_stride = 256 halves
                            0, 0};
                i32x4 gz = {0, 0, 0, 0};
#if defined(__clang_major__) && (__clang_major__ >= 23)
                i32x8 gz8 = {0, 0, 0, 0, 0, 0, 0, 0};
                __builtin_amdgcn_tensor_load_to_lds(g0, g1, gz, gz, gz8, 0);
#else
                __builtin_amdgcn_tensor_load_to_lds(g0, g1, gz, gz, 0);
#endif
            }
#else
            {
                const int row = tid >> 1;
                const int cs  = (tid & 1) * 16;
                const uint4* Ah = (const uint4*)((const _Float16*)A_ + (rbase + row) * DIMC + kc + cs);
                ((uint4*)&a_lds[row * 40 + cs])[0]     = Ah[0];
                ((uint4*)&a_lds[row * 40 + cs + 8])[0] = Ah[1];
            }
#endif
        } else {
            const int row = tid >> 1;
            const int cs  = (tid & 1) * 16;
            const float4* Af = (const float4*)((const float*)A_ + (rbase + row) * DIMC + kc + cs);
            float4 f0 = Af[0], f1 = Af[1], f2 = Af[2], f3 = Af[3];
            H8 h0, h1;
            h0.h[0] = (_Float16)f0.x; h0.h[1] = (_Float16)f0.y; h0.h[2] = (_Float16)f0.z; h0.h[3] = (_Float16)f0.w;
            h0.h[4] = (_Float16)f1.x; h0.h[5] = (_Float16)f1.y; h0.h[6] = (_Float16)f1.z; h0.h[7] = (_Float16)f1.w;
            h1.h[0] = (_Float16)f2.x; h1.h[1] = (_Float16)f2.y; h1.h[2] = (_Float16)f2.z; h1.h[3] = (_Float16)f2.w;
            h1.h[4] = (_Float16)f3.x; h1.h[5] = (_Float16)f3.y; h1.h[6] = (_Float16)f3.z; h1.h[7] = (_Float16)f3.w;
            ((uint4*)&a_lds[row * 40 + cs])[0]     = h0.u4;
            ((uint4*)&a_lds[row * 40 + cs + 8])[0] = h1.u4;
        }
        // ---- stage B tile 32x64 transposed: bt[col][k] = W[kc+k][cbase+col] ----
        {
            const int col = tid & 63;
            const int ks  = (tid >> 6) * 8;
            H8 hb;
#pragma unroll
            for (int e = 0; e < 8; ++e)
                hb.h[e] = (_Float16)W[(long)(kc + ks + e) * P + cbase + col];
            ((uint4*)&bt_lds[col * 40 + ks])[0] = hb.u4;
        }
#ifdef HAS_TDM
        if constexpr (A_F16) {
            if (wave == 0) __builtin_amdgcn_s_wait_tensorcnt(0);
        }
#endif
        __syncthreads();

        v16h a0 = frag_a_rowmajor(&a_lds[(wr + 0)  * 40], 40, lane);
        v16h a1 = frag_a_rowmajor(&a_lds[(wr + 16) * 40], 40, lane);
        v16h b0 = frag_b_kcontig(&bt_lds[(wc + 0)  * 40], 40, lane);
        v16h b1 = frag_b_kcontig(&bt_lds[(wc + 16) * 40], 40, lane);

        c[0][0] = wmma32(a0, b0, c[0][0]);
        c[0][1] = wmma32(a0, b1, c[0][1]);
        c[1][0] = wmma32(a1, b0, c[1][0]);
        c[1][1] = wmma32(a1, b1, c[1][1]);
        __syncthreads();
    }

    const int ln15 = lane & 15;
    const int hi8  = (lane & 16) ? 8 : 0;
#pragma unroll
    for (int tj = 0; tj < 2; ++tj) {
        const int col = cbase + wc + tj * 16 + ln15;
        const float bv = bias[col];
#pragma unroll
        for (int ti = 0; ti < 2; ++ti) {
#pragma unroll
            for (int r = 0; r < 8; ++r) {
                const long row = rbase + wr + ti * 16 + r + hi8;
                const float val = c[ti][tj][r] + bv;
                if (OUT_F32) ((float*)out_)[row * P + col] = val;
                else         ((_Float16*)out_)[row * P + col] = (_Float16)val;
            }
        }
    }
}

// ---------------- fused per-patch attention ----------------
// grid = MPATCH blocks, 256 threads = 8 waves, wave == head.
// Per-wave LDS region (7680 B): q[32][40] f16, k[32][40] f16, v[32][40] f16
// (all staged via global_load_async_to_lds_b128). After S: q/k region reused
// as S f32 (pitch 33), then as P f16 (pitch 40). Softmax scale folded into
// the f32 S epilogue, so staging is a pure byte copy.
__global__ __launch_bounds__(256) void attn_kernel(const _Float16* __restrict__ q16,
                                                   const _Float16* __restrict__ kv16,
                                                   const int*   __restrict__ rel_pos,
                                                   const float* __restrict__ maskp,
                                                   const float* __restrict__ rpe_table,
                                                   _Float16* __restrict__ o16) {
    __shared__ _Float16 rpe_l[RPE3 * NHEADS];                    // 2448 B
    __shared__ __align__(16) unsigned char wbuf[NHEADS][7680];   // 61440 B

    const int tid  = threadIdx.x;
    const int lane = tid & 31;
    const int h    = tid >> 5;
    const int m    = blockIdx.x;

    for (int i = tid; i < RPE3 * NHEADS; i += 256) rpe_l[i] = (_Float16)rpe_table[i];

    _Float16* q_l = (_Float16*)&wbuf[h][0];
    _Float16* k_l = (_Float16*)&wbuf[h][2560];
    _Float16* v_l = (_Float16*)&wbuf[h][5120];
    float*    s_l = (float*)&wbuf[h][0];
    _Float16* p_l = (_Float16*)&wbuf[h][0];

    // ---- async stage q, k, v tiles (one row of 32 halves per lane) ----
    {
        const long tok = (long)m * PK + lane;
        const _Float16* qg = q16  + tok * DIMC + h * HDIM;
        const _Float16* kg = kv16 + tok * (2 * DIMC) + h * HDIM;
        const _Float16* vg = kv16 + tok * (2 * DIMC) + DIMC + h * HDIM;
        const unsigned row_off = (unsigned)(lane * 40 * 2);
        async_row64((unsigned)(uintptr_t)q_l + row_off, (unsigned long long)(uintptr_t)qg);
        async_row64((unsigned)(uintptr_t)k_l + row_off, (unsigned long long)(uintptr_t)kg);
        async_row64((unsigned)(uintptr_t)v_l + row_off, (unsigned long long)(uintptr_t)vg);
        async_wait0();
    }
    __syncthreads();

    // ---- S = q @ k^T : 4 WMMAs (f32 accumulate) ----
    v16h aq[2], bk[2];
#pragma unroll
    for (int t = 0; t < 2; ++t) {
        aq[t] = frag_a_rowmajor(q_l + t * 16 * 40, 40, lane);
        bk[t] = frag_b_kcontig(k_l + t * 16 * 40, 40, lane);   // tileT[n=j][k=d]
    }
    v8f s[2][2];
#pragma unroll
    for (int ti = 0; ti < 2; ++ti)
#pragma unroll
        for (int tj = 0; tj < 2; ++tj) s[ti][tj] = wmma32(aq[ti], bk[tj], v8f_zero());

    // ---- scale + RPE bias + mask (rel_pos via b96 loads) ----
    const int ln15 = lane & 15;
    const int hi8  = (lane & 16) ? 8 : 0;
    float sreg[2][2][8];
#pragma unroll
    for (int ti = 0; ti < 2; ++ti)
#pragma unroll
        for (int tj = 0; tj < 2; ++tj)
#pragma unroll
            for (int r = 0; r < 8; ++r) {
                const int i = ti * 16 + r + hi8;
                const int j = tj * 16 + ln15;
                const long ij = ((long)m * PK + i) * PK + j;
                const I3 rp = *(const I3*)(rel_pos + ij * 3);
                int r0 = (rp.x < -POSB ? -POSB : (rp.x > POSB ? POSB : rp.x)) + POSB;
                int r1 = (rp.y < -POSB ? -POSB : (rp.y > POSB ? POSB : rp.y)) + POSB + RPEN;
                int r2 = (rp.z < -POSB ? -POSB : (rp.z > POSB ? POSB : rp.z)) + POSB + 2 * RPEN;
                const float bias = (float)rpe_l[r0 * NHEADS + h] + (float)rpe_l[r1 * NHEADS + h] +
                                   (float)rpe_l[r2 * NHEADS + h] + maskp[ij];
                sreg[ti][tj][r] = s[ti][tj][r] * SCALE_F + bias;
            }
    __syncthreads();   // q/k region dead; safe to overwrite with S
#pragma unroll
    for (int ti = 0; ti < 2; ++ti)
#pragma unroll
        for (int tj = 0; tj < 2; ++tj)
#pragma unroll
            for (int r = 0; r < 8; ++r)
                s_l[(ti * 16 + r + hi8) * 33 + tj * 16 + ln15] = sreg[ti][tj][r];
    __syncthreads();

    // ---- softmax: one lane per row ----
    {
        const int i = lane;
        float sv[PK];
        float mx = -1e30f;
#pragma unroll
        for (int j = 0; j < PK; ++j) { sv[j] = s_l[i * 33 + j]; mx = fmaxf(mx, sv[j]); }
        float sum = 0.0f;
#pragma unroll
        for (int j = 0; j < PK; ++j) { sv[j] = __expf(sv[j] - mx); sum += sv[j]; }
        const float inv = 1.0f / sum;
        __syncthreads();   // all rows read before P overwrites S
#pragma unroll
        for (int j = 0; j < PK; j += 2) {
            H2 pk;
            pk.h[0] = (_Float16)(sv[j] * inv);
            pk.h[1] = (_Float16)(sv[j + 1] * inv);
            *(unsigned*)&p_l[i * 40 + j] = pk.u;
        }
    }
    __syncthreads();

    // ---- O = P @ V : 4 WMMAs ----
    v16h ap[2], bv[2];
#pragma unroll
    for (int t = 0; t < 2; ++t) {
        ap[t] = frag_a_rowmajor(p_l + t * 16 * 40, 40, lane);
        bv[t] = frag_b_nmajor(v_l + t * 16, 40, lane);   // row-major v[j][d]
    }
    v8f o[2][2];
#pragma unroll
    for (int ti = 0; ti < 2; ++ti)
#pragma unroll
        for (int tj = 0; tj < 2; ++tj) o[ti][tj] = wmma32(ap[ti], bv[tj], v8f_zero());

#pragma unroll
    for (int ti = 0; ti < 2; ++ti)
#pragma unroll
        for (int tj = 0; tj < 2; ++tj)
#pragma unroll
            for (int r = 0; r < 8; ++r) {
                const int i = ti * 16 + r + hi8;
                const int d = tj * 16 + ln15;
                o16[((long)m * PK + i) * DIMC + h * HDIM + d] = (_Float16)o[ti][tj][r];
            }
}

// ---------------- launch ----------------
extern "C" void kernel_launch(void* const* d_in, const int* in_sizes, int n_in,
                              void* d_out, int out_size, void* d_ws, size_t ws_size,
                              hipStream_t stream) {
    const float* q_data    = (const float*)d_in[0];
    const float* kv_data   = (const float*)d_in[1];
    const int*   rel_pos   = (const int*)  d_in[2];
    const float* mask      = (const float*)d_in[3];
    const float* Wq        = (const float*)d_in[4];
    const float* bq        = (const float*)d_in[5];
    const float* Wkv       = (const float*)d_in[6];
    const float* bkv       = (const float*)d_in[7];
    const float* Wo        = (const float*)d_in[8];
    const float* bo        = (const float*)d_in[9];
    const float* rpe_table = (const float*)d_in[10];
    float* out = (float*)d_out;

    _Float16* q16  = (_Float16*)d_ws;                      // N x 256 f16
    _Float16* kv16 = q16  + (size_t)NPTS * DIMC;           // N x 512 f16
    _Float16* o16  = kv16 + (size_t)NPTS * 2 * DIMC;       // N x 256 f16

    dim3 blk(256);
    gemm_kernel<false, false><<<dim3(DIMC / 64, NPTS / 128), blk, 0, stream>>>(
        (const void*)q_data, Wq, bq, (void*)q16, DIMC);
    gemm_kernel<false, false><<<dim3(2 * DIMC / 64, NPTS / 128), blk, 0, stream>>>(
        (const void*)kv_data, Wkv, bkv, (void*)kv16, 2 * DIMC);
    attn_kernel<<<dim3(MPATCH), blk, 0, stream>>>(q16, kv16, rel_pos, mask, rpe_table, o16);
    gemm_kernel<true, true><<<dim3(DIMC / 64, NPTS / 128), blk, 0, stream>>>(
        (const void*)o16, Wo, bo, (void*)out, DIMC);
}
